// PointNet2Decoder_14078902796574
// MI455X (gfx1250) — compile-verified, matrix operand order checked by reading, not run on hardware
//
#include <hip/hip_runtime.h>
#include <hip/hip_bf16.h>

// ---------------- types ----------------
typedef _Float16 f16;
typedef __attribute__((ext_vector_type(16))) _Float16 v16h;
typedef __attribute__((ext_vector_type(8)))  _Float16 v8h;
typedef __attribute__((ext_vector_type(8)))  float    v8f;
typedef __attribute__((ext_vector_type(4)))  unsigned int v4u;
typedef __attribute__((ext_vector_type(4)))  int v4i;
typedef __attribute__((ext_vector_type(8)))  int v8i;

#ifndef __has_builtin
#define __has_builtin(x) 0
#endif
#if __has_builtin(__builtin_amdgcn_tensor_load_to_lds) && \
    __has_builtin(__builtin_amdgcn_s_wait_tensorcnt)
#define HAVE_TDM 1
#else
#define HAVE_TDM 0
#endif

static __device__ __forceinline__ v8f wmma_f16(v16h a, v16h b, v8f c) {
  // D = A(16x32 f16) x B(32x16 f16) + C(16x16 f32)
  return __builtin_amdgcn_wmma_f32_16x16x32_f16(
      /*neg_a=*/false, a, /*neg_b=*/false, b,
      /*c_mod=*/(short)0, c, /*reuse_a=*/false, /*reuse_b=*/false);
}

// A fragment: row-major f16 matrix (global or LDS), 16 rows at A0, ld=ldk,
// K-block at kb. Layout (ISA 7.12.2, 16-bit A 16x32): lanes m / m+16 hold row m;
// lanes 0-15: K=kb+0..7, kb+16..23 ; lanes 16-31: K=kb+8..15, kb+24..31.
static __device__ __forceinline__ v16h load_a_frag(const f16* __restrict__ A0,
                                                   int ldk, int kb) {
  int lane = threadIdx.x & 31;
  int m = lane & 15, hi = lane >> 4;
  const f16* p = A0 + (long)m * ldk + kb + hi * 8;
  union { v16h v; v8h h[2]; } u;
  u.h[0] = *(const v8h*)(p);
  u.h[1] = *(const v8h*)(p + 16);
  return u.v;
}

// B fragment from pre-packed buffer: fragment f is 32 lanes x 16 halves contiguous.
static __device__ __forceinline__ v16h load_b_frag(const f16* __restrict__ Bp,
                                                   long fragIdx) {
  int lane = threadIdx.x & 31;
  return *(const v16h*)(Bp + (fragIdx * 32 + lane) * 16);
}

#if HAVE_TDM
// 1-D TDM copy: n2B f16 elements from global -> LDS (wave-scalar op, EXEC ignored).
static __device__ __forceinline__ void tdm_copy_1d(unsigned int lds_addr,
                                                   const void* gaddr,
                                                   unsigned int n2B) {
  unsigned long long ga = (unsigned long long)gaddr;
  v4u g0 = {};
  g0.x = 1u;                                   // count=1 (valid user descriptor)
  g0.y = lds_addr;                             // lds_addr (bytes)
  g0.z = (unsigned int)ga;                     // global_addr[31:0]
  g0.w = ((unsigned int)(ga >> 32) & 0x01FFFFFFu) | (2u << 30); // addr[56:32], type=2
  v8i g1 = {};
  g1[0] = (1 << 16);                           // data_size = 2 bytes
  g1[1] = (int)((n2B & 0xFFFFu) << 16);        // tensor_dim0[15:0]
  g1[2] = (int)(((n2B >> 16) & 0xFFFFu) | (1u << 16)); // tensor_dim0[31:16], tensor_dim1=1
  g1[3] = (int)((n2B & 0xFFFFu) << 16);        // tile_dim0 (<=65535)
  g1[4] = 0;                                   // tile_dim1/2 unused (1-D)
  g1[5] = (int)n2B;                            // tensor_dim0_stride
  g1[6] = 0;
  g1[7] = 0;
  v4i z4 = {};
#if defined(__clang_major__) && (__clang_major__ >= 23)
  v8i z8 = {};
  __builtin_amdgcn_tensor_load_to_lds(g0, g1, z4, z4, z8, 0);
#else
  __builtin_amdgcn_tensor_load_to_lds(g0, g1, z4, z4, 0);
#endif
}
#endif

// async global -> LDS 16-byte copy, per-lane addresses (ASYNCcnt-tracked)
static __device__ __forceinline__ void async_copy_b128(unsigned int lds_addr,
                                                       const void* base,
                                                       unsigned int byte_off) {
  asm volatile("global_load_async_to_lds_b128 %0, %1, %2"
               :: "v"(lds_addr), "v"(byte_off), "s"(base) : "memory");
}
static __device__ __forceinline__ void wait_asynccnt0() {
  asm volatile("s_wait_asynccnt 0x0" ::: "memory");
}

// ---------------- weight packing: f32 [K,N] row-major -> f16 B fragments ---
__global__ void pack_b_kernel(const float* __restrict__ W, f16* __restrict__ Bp,
                              int K, int N) {
  int kb32 = K >> 5;
  long total = (long)K * N;
  for (long i = (long)blockIdx.x * blockDim.x + threadIdx.x; i < total;
       i += (long)gridDim.x * blockDim.x) {
    long frag = i >> 9;         // /512
    int  rem  = (int)(i & 511);
    int  lane = rem >> 4;
    int  h    = rem & 15;
    int  ct   = (int)(frag / kb32);
    int  kb   = (int)(frag % kb32);
    int  hi   = lane >> 4, n = lane & 15;
    int  k    = kb * 32 + hi * 8 + (h < 8 ? h : h + 8);
    int  col  = ct * 16 + n;
    Bp[i] = (f16)W[(long)k * N + col];
  }
}

// ---------------- small helpers ----------------
__global__ void zero_f32_kernel(float* p, int n) {
  int i = blockIdx.x * blockDim.x + threadIdx.x;
  if (i < n) p[i] = 0.f;
}
__global__ void f32_to_f16_kernel(const float* __restrict__ s, f16* __restrict__ d, int n) {
  int i = blockIdx.x * blockDim.x + threadIdx.x;
  if (i < n) d[i] = (f16)s[i];
}
__global__ void gather3_kernel(const float* __restrict__ src, const int* __restrict__ idx,
                               float* __restrict__ dst, int Npts, int S, int total) {
  int i = blockIdx.x * blockDim.x + threadIdx.x;
  if (i >= total) return;
  int c = i % 3, bs = i / 3, b = bs / S;
  dst[i] = src[((long)b * Npts + idx[bs]) * 3 + c];
}

// ---------------- STN point MLP 3->64->128 (f16 out) ----------------
__global__ void stn_mlp12_kernel(const float* __restrict__ xyz,
                                 const float* __restrict__ w1,
                                 const float* __restrict__ w2,
                                 f16* __restrict__ h2) {
  int p = blockIdx.x;                 // one point per block, 128 threads
  int t = threadIdx.x;
  __shared__ float sx[3];
  __shared__ float sh1[64];
  if (t < 3) sx[t] = xyz[(long)p * 3 + t];
  __syncthreads();
  if (t < 64) {
    float a = sx[0] * w1[t] + sx[1] * w1[64 + t] + sx[2] * w1[128 + t];
    sh1[t] = fmaxf(a, 0.f);
  }
  __syncthreads();
  float a = 0.f;
  #pragma unroll 8
  for (int k = 0; k < 64; ++k) a += sh1[k] * w2[k * 128 + t];
  h2[(long)p * 128 + t] = (f16)fmaxf(a, 0.f);
}

// ---------------- STN big GEMM 128->1024 + global max-pool ----------------
// block = (b, rowtile of 16 points), 8 waves, wave covers 8 column tiles.
// A tile (16x128 f16 = 4KB, contiguous) staged into LDS via the TDM.
__global__ void stn_gemm_pool_kernel(const f16* __restrict__ h2,
                                     const f16* __restrict__ Wp,
                                     float* __restrict__ pooled) {
  __shared__ f16 sA[16 * 128];
  int blk = blockIdx.x;
  int b = blk >> 8, rt = blk & 255;
  int wave = threadIdx.x >> 5, lane = threadIdx.x & 31;
  const f16* A0 = h2 + ((long)b * 4096 + (long)rt * 16) * 128;
#if HAVE_TDM
  if (threadIdx.x < 32) {   // wave 0 drives the TDM (scalar op; EXEC ignored)
    tdm_copy_1d((unsigned int)(uintptr_t)sA, A0, 16 * 128);
    __builtin_amdgcn_s_wait_tensorcnt(0);
  }
  __syncthreads();
#else
  for (int i = threadIdx.x; i < 256; i += 256)
    ((v8h*)sA)[i] = ((const v8h*)A0)[i];
  __syncthreads();
#endif
  v8f c[8] = {};
  for (int kb = 0; kb < 4; ++kb) {
    v16h a = load_a_frag(sA, 128, kb * 32);
    v16h bf[8];
    #pragma unroll
    for (int j = 0; j < 8; ++j)   // issue all B loads, then let waits pipeline
      bf[j] = load_b_frag(Wp, (long)(wave * 8 + j) * 4 + kb);
    #pragma unroll
    for (int j = 0; j < 8; ++j)
      c[j] = wmma_f16(a, bf[j], c[j]);
  }
  int n = lane & 15;
  #pragma unroll
  for (int j = 0; j < 8; ++j) {
    float m = c[j][0];
    #pragma unroll
    for (int r = 1; r < 8; ++r) m = fmaxf(m, c[j][r]);
    m = fmaxf(m, __shfl_xor(m, 16, 32));    // combine row halves
    m = fmaxf(m, 0.f);                       // relu (pool init is 0)
    if (lane < 16)
      atomicMax(reinterpret_cast<int*>(pooled + b * 1024 + (wave * 8 + j) * 16 + n),
                __float_as_int(m));
  }
}

// ---------------- generic 16-row-tile WMMA GEMM (relu, optional bias/f16 out)
__global__ void gemm16_kernel(const f16* __restrict__ A, const f16* __restrict__ Bp,
                              const float* __restrict__ bias,
                              float* __restrict__ C, f16* __restrict__ C16,
                              int M, int N, int K, int relu) {
  int gw = (blockIdx.x * blockDim.x + threadIdx.x) >> 5;
  int nct = N >> 4;
  int ntiles = (M >> 4) * nct;
  if (gw >= ntiles) return;                  // wave-uniform
  int rt = gw / nct, ct = gw % nct;
  int lane = threadIdx.x & 31;
  int kb32 = K >> 5;
  const f16* A0 = A + (long)rt * 16 * K;
  v8f acc = {};
  for (int kb = 0; kb < kb32; ++kb) {
    __builtin_prefetch(A0 + (kb + 1) * 32, 0, 1);
    v16h a  = load_a_frag(A0, K, kb * 32);
    v16h bf = load_b_frag(Bp, (long)ct * kb32 + kb);
    acc = wmma_f16(a, bf, acc);
  }
  int hi = lane >> 4, n = lane & 15, col = ct * 16 + n;
  float bv = bias ? bias[col] : 0.f;
  #pragma unroll
  for (int r = 0; r < 8; ++r) {
    int row = rt * 16 + hi * 8 + r;
    float v = acc[r] + bv;
    if (relu) v = fmaxf(v, 0.f);
    if (C)   C[(long)row * N + col] = v;
    if (C16) C16[(long)row * N + col] = (f16)v;
  }
}

// ---------------- STN fc3 (256->9) + identity offset ----------------
__global__ void stn_fc3_kernel(const float* __restrict__ h, const float* __restrict__ f3,
                               float* __restrict__ trans) {
  int i = blockIdx.x * blockDim.x + threadIdx.x;
  if (i >= 16 * 9) return;
  int b = i / 9, j = i % 9;
  float a = 0.f;
  for (int k = 0; k < 256; ++k) a += h[b * 256 + k] * f3[k * 9 + j];
  if (j == 0 || j == 4 || j == 8) a += 1.f;
  trans[i] = a;
}

// ---------------- apply 3x3 transform ----------------
__global__ void transform_kernel(const float* __restrict__ xyz, const float* __restrict__ trans,
                                 float* __restrict__ xyz_t) {
  int p = blockIdx.x * blockDim.x + threadIdx.x;
  if (p >= 16 * 4096) return;
  int b = p >> 12;
  float x = xyz[(long)p * 3], y = xyz[(long)p * 3 + 1], z = xyz[(long)p * 3 + 2];
  const float* T = trans + b * 9;
  xyz_t[(long)p * 3 + 0] = x * T[0] + y * T[3] + z * T[6];
  xyz_t[(long)p * 3 + 1] = x * T[1] + y * T[4] + z * T[7];
  xyz_t[(long)p * 3 + 2] = x * T[2] + y * T[5] + z * T[8];
}

// ---------------- farthest point sampling (one workgroup per batch) ------
__global__ void fps_kernel(const float* __restrict__ xyz, int Npts, int npoint,
                           int* __restrict__ outIdx) {
  int b = blockIdx.x;
  const float* P = xyz + (long)b * Npts * 3;
  int t = threadIdx.x;
  float d[16];
  {
    int cnt = 0;
    for (int i = t; i < Npts; i += 256) d[cnt++] = 1e10f;
  }
  __shared__ float sc[3];
  __shared__ int s_far;
  __shared__ float s_val[256];
  __shared__ int s_idx[256];
  if (t == 0) s_far = 0;
  __syncthreads();
  for (int it = 0; it < npoint; ++it) {
    int far = s_far;
    if (t == 0) {
      outIdx[(long)b * npoint + it] = far;
      sc[0] = P[(long)far * 3]; sc[1] = P[(long)far * 3 + 1]; sc[2] = P[(long)far * 3 + 2];
    }
    __syncthreads();
    float cx = sc[0], cy = sc[1], cz = sc[2];
    float best = -1.f; int bi = 0; int cnt = 0;
    for (int i = t; i < Npts; i += 256) {
      float dx = P[(long)i * 3] - cx, dy = P[(long)i * 3 + 1] - cy, dz = P[(long)i * 3 + 2] - cz;
      float nd = fminf(d[cnt], dx * dx + dy * dy + dz * dz);
      d[cnt++] = nd;
      if (nd > best) { best = nd; bi = i; }
    }
    s_val[t] = best; s_idx[t] = bi;
    __syncthreads();
    for (int off = 128; off > 0; off >>= 1) {
      if (t < off && s_val[t + off] > s_val[t]) { s_val[t] = s_val[t + off]; s_idx[t] = s_idx[t + off]; }
      __syncthreads();
    }
    if (t == 0) s_far = s_idx[0];
    __syncthreads();
  }
}

// ---------------- KNN via iterative selection (distances cached in LDS) --
__global__ void knn_kernel(const float* __restrict__ q, const float* __restrict__ ref,
                           int S, int Npts, int nsample, int* __restrict__ outIdx) {
  int bs = blockIdx.x;
  int b = bs / S;
  const float* R = ref + (long)b * Npts * 3;
  float qx = q[(long)bs * 3], qy = q[(long)bs * 3 + 1], qz = q[(long)bs * 3 + 2];
  __shared__ float s_d[4096];
  __shared__ float s_val[256];
  __shared__ int s_idx[256];
  int t = threadIdx.x;
  for (int i = t; i < Npts; i += 256) {
    float dx = R[(long)i * 3] - qx, dy = R[(long)i * 3 + 1] - qy, dz = R[(long)i * 3 + 2] - qz;
    s_d[i] = dx * dx + dy * dy + dz * dz;
  }
  __syncthreads();
  for (int k = 0; k < nsample; ++k) {
    float best = 3e38f; int bi = 0;
    for (int i = t; i < Npts; i += 256) {
      float v = s_d[i];
      if (v < best) { best = v; bi = i; }
    }
    s_val[t] = best; s_idx[t] = bi;
    __syncthreads();
    for (int off = 128; off > 0; off >>= 1) {
      if (t < off && s_val[t + off] < s_val[t]) { s_val[t] = s_val[t + off]; s_idx[t] = s_idx[t + off]; }
      __syncthreads();
    }
    if (t == 0) { outIdx[(long)bs * nsample + k] = s_idx[0]; s_d[s_idx[0]] = 3e38f; }
    __syncthreads();
  }
}

// ---------------- SA1 grouped MLP 3->128 + max over 32 (scalar; K=3) -----
__global__ void sa1_kernel(const float* __restrict__ xyz_t, const int* __restrict__ nidx,
                           const float* __restrict__ w, const float* __restrict__ bias,
                           f16* __restrict__ l1h) {
  int bs = blockIdx.x;           // b*512+s, 128 threads
  int b = bs >> 9;
  __shared__ float g[32][3];
  int t = threadIdx.x;
  if (t < 96) {
    int gi = t / 3, c = t % 3;
    int src = nidx[(long)bs * 32 + gi];
    g[gi][c] = xyz_t[((long)b * 4096 + src) * 3 + c];
  }
  __syncthreads();
  float w0 = w[t], w1c = w[128 + t], w2c = w[256 + t], bj = bias[t];
  float m = 0.f;   // max of relu values (>=0)
  #pragma unroll 8
  for (int k = 0; k < 32; ++k) {
    float v = g[k][0] * w0 + g[k][1] * w1c + g[k][2] * w2c + bj;
    m = fmaxf(m, v);
  }
  l1h[(long)bs * 128 + t] = (f16)fmaxf(m, 0.f);
}

// ---------------- SA2 grouped GEMM 128->256 + max over 64 (WMMA) ---------
__global__ void sa2_kernel(const f16* __restrict__ l1h, const int* __restrict__ nidx,
                           const f16* __restrict__ Wp, const float* __restrict__ bias,
                           f16* __restrict__ l2h) {
  int bs = blockIdx.x;            // b*128+s, 256 threads = 8 waves
  int b = bs >> 7;
  int wave = threadIdx.x >> 5, lane = threadIdx.x & 31;
  __shared__ int sIdx[64];
  if (threadIdx.x < 64) sIdx[threadIdx.x] = nidx[(long)bs * 64 + threadIdx.x];
  __syncthreads();
  int m = lane & 15, hi = lane >> 4;
  #pragma unroll
  for (int j = 0; j < 2; ++j) {
    int ct = wave * 2 + j;
    float colmax = -3e38f;
    for (int rt = 0; rt < 4; ++rt) {
      v8f c = {};
      const f16* arow = l1h + ((long)b * 512 + sIdx[rt * 16 + m]) * 128;
      #pragma unroll
      for (int kb = 0; kb < 4; ++kb) {
        union { v16h v; v8h h[2]; } u;
        u.h[0] = *(const v8h*)(arow + kb * 32 + hi * 8);
        u.h[1] = *(const v8h*)(arow + kb * 32 + 16 + hi * 8);
        v16h bf = load_b_frag(Wp, (long)ct * 4 + kb);
        c = wmma_f16(u.v, bf, c);
      }
      #pragma unroll
      for (int r = 0; r < 8; ++r) colmax = fmaxf(colmax, c[r]);
    }
    colmax = fmaxf(colmax, __shfl_xor(colmax, 16, 32));
    int col = ct * 16 + (lane & 15);
    float v = fmaxf(colmax + bias[col], 0.f);
    if (lane < 16) l2h[(long)bs * 256 + col] = (f16)v;
  }
}

// ---------------- l3: relu(l2 @ W + b) max over 128 rows (WMMA) ----------
// block = batch (512 thr, 16 waves x 4 coltiles); A (128x256 f16 = 64KB)
// staged in LDS via async global->LDS loads so global A traffic is 1x.
__global__ void l3_kernel(const f16* __restrict__ l2h, const f16* __restrict__ Wp,
                          const float* __restrict__ bias, f16* __restrict__ l3h) {
  __shared__ f16 sA[128 * 256];    // 64KB
  int b = blockIdx.x;
  const f16* A = l2h + (long)b * 128 * 256;
  unsigned int lds0 = (unsigned int)(uintptr_t)sA;
  for (int i = threadIdx.x; i < 4096; i += 512)      // 4096 x 16B = 64KB
    async_copy_b128(lds0 + i * 16, A, (unsigned int)(i * 16));
  wait_asynccnt0();
  __syncthreads();
  int wave = threadIdx.x >> 5, lane = threadIdx.x & 31;
  #pragma unroll
  for (int j = 0; j < 4; ++j) {
    int ct = wave * 4 + j;
    float colmax = -3e38f;
    for (int rt = 0; rt < 8; ++rt) {
      v8f c = {};
      const f16* A0 = sA + rt * 16 * 256;
      #pragma unroll
      for (int kb = 0; kb < 8; ++kb) {
        v16h a  = load_a_frag(A0, 256, kb * 32);
        v16h bf = load_b_frag(Wp, (long)ct * 8 + kb);
        c = wmma_f16(a, bf, c);
      }
      #pragma unroll
      for (int r = 0; r < 8; ++r) colmax = fmaxf(colmax, c[r]);
    }
    colmax = fmaxf(colmax, __shfl_xor(colmax, 16, 32));
    int col = ct * 16 + (lane & 15);
    float v = fmaxf(colmax + bias[col], 0.f);
    if (lane < 16) l3h[(long)b * 1024 + col] = (f16)v;
  }
}

// ======================================================================
extern "C" void kernel_launch(void* const* d_in, const int* in_sizes, int n_in,
                              void* d_out, int out_size, void* d_ws, size_t ws_size,
                              hipStream_t stream) {
  (void)in_sizes; (void)n_in; (void)out_size; (void)ws_size;
  const float* xyz  = (const float*)d_in[0];
  const float* w1   = (const float*)d_in[1];
  const float* w2   = (const float*)d_in[2];
  const float* w3   = (const float*)d_in[3];
  const float* f1   = (const float*)d_in[4];
  const float* f2   = (const float*)d_in[5];
  const float* f3   = (const float*)d_in[6];
  const float* sa1w = (const float*)d_in[7];
  const float* sa1b = (const float*)d_in[8];
  const float* sa2w = (const float*)d_in[9];
  const float* sa2b = (const float*)d_in[10];
  const float* sa3w = (const float*)d_in[11];
  const float* sa3b = (const float*)d_in[12];
  const float* finw = (const float*)d_in[13];
  const float* finb = (const float*)d_in[14];
  float* out = (float*)d_out;

  char* wsb = (char*)d_ws;
  auto alloc = [&](size_t bytes) -> void* {
    void* p = (void*)wsb;
    wsb += (bytes + 255) & ~(size_t)255;
    return p;
  };
  const int B = 16, N = 4096;
  f16*   h2       = (f16*)alloc((size_t)B * N * 128 * 2);
  f16*   w3p      = (f16*)alloc((size_t)128 * 1024 * 2);
  f16*   f1p      = (f16*)alloc((size_t)1024 * 512 * 2);
  f16*   f2p      = (f16*)alloc((size_t)512 * 256 * 2);
  f16*   sa2p     = (f16*)alloc((size_t)128 * 256 * 2);
  f16*   sa3p     = (f16*)alloc((size_t)256 * 1024 * 2);
  f16*   finp     = (f16*)alloc((size_t)1024 * 256 * 2);
  float* pooled   = (float*)alloc((size_t)B * 1024 * 4);
  f16*   pooled16 = (f16*)alloc((size_t)B * 1024 * 2);
  f16*   fc1h     = (f16*)alloc((size_t)B * 512 * 2);
  float* fc2f     = (float*)alloc((size_t)B * 256 * 4);
  float* trans    = (float*)alloc((size_t)B * 9 * 4);
  float* xyz_t    = (float*)alloc((size_t)B * N * 3 * 4);
  int*   idx1     = (int*)alloc((size_t)B * 512 * 4);
  float* l1xyz    = (float*)alloc((size_t)B * 512 * 3 * 4);
  int*   knn1     = (int*)alloc((size_t)B * 512 * 32 * 4);
  f16*   l1h      = (f16*)alloc((size_t)B * 512 * 128 * 2);
  int*   idx2     = (int*)alloc((size_t)B * 128 * 4);
  float* l2xyz    = (float*)alloc((size_t)B * 128 * 3 * 4);
  int*   knn2     = (int*)alloc((size_t)B * 128 * 64 * 4);
  f16*   l2h      = (f16*)alloc((size_t)B * 128 * 256 * 2);
  f16*   l3h      = (f16*)alloc((size_t)B * 1024 * 2);

  // --- pack all WMMA weights (runs every launch; trivial cost) ---
  pack_b_kernel<<<512, 256, 0, stream>>>(w3, w3p, 128, 1024);
  pack_b_kernel<<<2048, 256, 0, stream>>>(f1, f1p, 1024, 512);
  pack_b_kernel<<<512, 256, 0, stream>>>(f2, f2p, 512, 256);
  pack_b_kernel<<<128, 256, 0, stream>>>(sa2w, sa2p, 128, 256);
  pack_b_kernel<<<1024, 256, 0, stream>>>(sa3w, sa3p, 256, 1024);
  pack_b_kernel<<<1024, 256, 0, stream>>>(finw, finp, 1024, 256);

  // --- STN ---
  stn_mlp12_kernel<<<B * N, 128, 0, stream>>>(xyz, w1, w2, h2);
  zero_f32_kernel<<<64, 256, 0, stream>>>(pooled, B * 1024);
  stn_gemm_pool_kernel<<<B * 256, 256, 0, stream>>>(h2, w3p, pooled);
  f32_to_f16_kernel<<<64, 256, 0, stream>>>(pooled, pooled16, B * 1024);
  gemm16_kernel<<<8, 128, 0, stream>>>(pooled16, f1p, nullptr, nullptr, fc1h,
                                       16, 512, 1024, 1);
  gemm16_kernel<<<4, 128, 0, stream>>>(fc1h, f2p, nullptr, fc2f, nullptr,
                                       16, 256, 512, 1);
  stn_fc3_kernel<<<1, 160, 0, stream>>>(fc2f, f3, trans);
  transform_kernel<<<256, 256, 0, stream>>>(xyz, trans, xyz_t);

  // --- SA1: FPS(512) -> KNN(32) -> MLP 3->128 -> maxpool ---
  fps_kernel<<<B, 256, 0, stream>>>(xyz_t, N, 512, idx1);
  gather3_kernel<<<96, 256, 0, stream>>>(xyz_t, idx1, l1xyz, N, 512, B * 512 * 3);
  knn_kernel<<<B * 512, 256, 0, stream>>>(l1xyz, xyz_t, 512, N, 32, knn1);
  sa1_kernel<<<B * 512, 128, 0, stream>>>(xyz_t, knn1, sa1w, sa1b, l1h);

  // --- SA2: FPS(128) -> KNN(64) -> MLP 128->256 (WMMA) -> maxpool ---
  fps_kernel<<<B, 256, 0, stream>>>(l1xyz, 512, 128, idx2);
  gather3_kernel<<<24, 256, 0, stream>>>(l1xyz, idx2, l2xyz, 512, 128, B * 128 * 3);
  knn_kernel<<<B * 128, 256, 0, stream>>>(l2xyz, l1xyz, 128, 512, 64, knn2);
  sa2_kernel<<<B * 128, 256, 0, stream>>>(l1h, knn2, sa2p, sa2b, l2h);

  // --- l3 global pool (WMMA, LDS-staged A) + final projection (WMMA) ---
  l3_kernel<<<B, 512, 0, stream>>>(l2h, sa3p, sa3b, l3h);
  gemm16_kernel<<<4, 128, 0, stream>>>(l3h, finp, finb, out, nullptr,
                                       16, 256, 1024, 0);
}
